// net_pgnn_84524956385829
// MI455X (gfx1250) — compile-verified, matrix operand order checked by reading, not run on hardware
//
#include <hip/hip_runtime.h>

// ---------------------------------------------------------------------------
// CDNA5 (gfx1250) implementation of the pGNN reference.
// Dense GEMMs use v_wmma_f32_16x16x32_f16 (wave32), with 2x4 register tile
// blocking per wave (32x64 output block) so each K-step issues 12 b128 loads
// for 8 WMMAs (vs 4 loads per WMMA unblocked) -- ~2.7x less L2 traffic.
// Sparse aggregation uses global f32 atomics (L2-resident, 340K edges).
// With P=2 the p-Laplacian weight M == w exactly (gnorm^0 == 1), so the
// per-edge gradient-norm pass is skipped (identical math).
// ---------------------------------------------------------------------------

typedef __attribute__((ext_vector_type(16))) _Float16 v16h;
typedef __attribute__((ext_vector_type(8)))  _Float16 v8h;
typedef __attribute__((ext_vector_type(8)))  float    v8f;

#define NUM_NODES 20000
#define NUM_EDGES 320000
#define IN_DIM    512
#define HID_DIM   512
#define OUT_DIM   128
#define MU_C      0.1f

#define MB 2   // M-tiles per wave
#define NB 4   // N-tiles per wave

// ---------------------------------------------------------------- utilities
__global__ void fill_f32_kernel(float* __restrict__ p, float v, long n) {
    long i = (long)blockIdx.x * blockDim.x + threadIdx.x;
    if (i < n) p[i] = v;
}

__global__ void f32_to_f16_kernel(const float* __restrict__ in,
                                  _Float16* __restrict__ out, long n) {
    long i = (long)blockIdx.x * blockDim.x + threadIdx.x;
    if (i < n) out[i] = (_Float16)in[i];
}

// W [K,Nc] f32 row-major  ->  Wt [Nc,K] f16 row-major; optional scale+relu
__global__ void transpose_w_kernel(const float* __restrict__ W,
                                   _Float16* __restrict__ Wt,
                                   int K, int Nc, float scale, int relu) {
    long idx = (long)blockIdx.x * blockDim.x + threadIdx.x;
    if (idx >= (long)K * Nc) return;
    int k = (int)(idx / Nc);
    int n = (int)(idx % Nc);
    float v = W[idx] * scale;
    if (relu) v = fmaxf(v, 0.0f);
    Wt[(long)n * K + k] = (_Float16)v;
}

// --------------------------------------------------------------- WMMA GEMM
// C[M,Nc] = A[M,K](f16) * Bt[Nc,K](f16, B transposed) + bias.
// Each wave owns a (MB*16) x (NB*16) output block; K stepped by 32 through
// v_wmma_f32_16x16x32_f16 with A/B fragments reused across the block.
__global__ __launch_bounds__(256)
void gemm_f16_wmma_kernel(const _Float16* __restrict__ A,
                          const _Float16* __restrict__ Bt,
                          const float* __restrict__ bias,
                          float* __restrict__ Cf32,
                          _Float16* __restrict__ Cf16,
                          int M, int K, int Nc, int relu) {
    const int lane = threadIdx.x & 31;
    const int wave = (blockIdx.x * blockDim.x + threadIdx.x) >> 5;
    const int gm = (M >> 4) / MB;                 // 1250/2 = 625
    const int gn = (Nc >> 4) / NB;                // tilesN/4 (all Nc here: /16 %4==0)
    if (wave >= gm * gn) return;                  // wave-uniform: EXEC stays ~0
    const int tm0 = (wave / gn) * MB;
    const int tn0 = (wave % gn) * NB;

    const int half16 = lane >> 4;                 // 0 | 1
    const int l15    = lane & 15;
    const int kk     = half16 << 3;               // 0 | 8  (ISA A/B K-offset; == row offset)

    long arow[MB], brow[NB];
#pragma unroll
    for (int mi = 0; mi < MB; ++mi)
        arow[mi] = (long)((tm0 + mi) * 16 + l15) * K;
#pragma unroll
    for (int ni = 0; ni < NB; ++ni)
        brow[ni] = (long)((tn0 + ni) * 16 + l15) * K;

    v8f acc[MB][NB] = {};
    for (int k0 = 0; k0 < K; k0 += 32) {
        union Frag { v16h v; v8h h[2]; };
        Frag a[MB], b[NB];
#pragma unroll
        for (int mi = 0; mi < MB; ++mi) {
            a[mi].h[0] = *(const v8h*)(A + arow[mi] + k0 + kk);
            a[mi].h[1] = *(const v8h*)(A + arow[mi] + k0 + 16 + kk);
        }
#pragma unroll
        for (int ni = 0; ni < NB; ++ni) {
            b[ni].h[0] = *(const v8h*)(Bt + brow[ni] + k0 + kk);
            b[ni].h[1] = *(const v8h*)(Bt + brow[ni] + k0 + 16 + kk);
        }
#pragma unroll
        for (int mi = 0; mi < MB; ++mi)
#pragma unroll
            for (int ni = 0; ni < NB; ++ni)
                acc[mi][ni] = __builtin_amdgcn_wmma_f32_16x16x32_f16(
                    false, a[mi].v, false, b[ni].v, (short)0, acc[mi][ni],
                    false, false);
    }

#pragma unroll
    for (int ni = 0; ni < NB; ++ni) {
        const int col = (tn0 + ni) * 16 + l15;
        const float bv = bias ? bias[col] : 0.0f;
#pragma unroll
        for (int mi = 0; mi < MB; ++mi) {
#pragma unroll
            for (int v = 0; v < 8; ++v) {
                int row = (tm0 + mi) * 16 + v + kk;   // C/D layout: M = v + 8*half16
                float r = acc[mi][ni][v] + bv;
                if (relu) r = fmaxf(r, 0.0f);
                if (Cf32) Cf32[(long)row * Nc + col] = r;
                if (Cf16) Cf16[(long)row * Nc + col] = (_Float16)r;
            }
        }
    }
}

// ------------------------------------------------------------ edge weights
// ew[e] = dot(logits[row[e]], tP[col[e]])   (tP = logits @ parsing, per node)
__global__ void edge_dot_kernel(const float* __restrict__ logits,
                                const float* __restrict__ tP,
                                const int* __restrict__ row,
                                const int* __restrict__ col,
                                float* __restrict__ ew, int E) {
    int e = blockIdx.x * blockDim.x + threadIdx.x;
    if (e >= E) return;
    const float4* a = (const float4*)(logits + (long)row[e] * OUT_DIM);
    const float4* b = (const float4*)(tP     + (long)col[e] * OUT_DIM);
    float s = 0.0f;
#pragma unroll
    for (int i = 0; i < OUT_DIM / 4; ++i) {
        float4 av = a[i], bv = b[i];
        s += av.x * bv.x + av.y * bv.y + av.z * bv.z + av.w * bv.w;
    }
    ew[e] = s;
}

__global__ void reduce_stats_kernel(const float* __restrict__ ew,
                                    float* __restrict__ stats, int E) {
    __shared__ float s1[256], s2[256];
    int tid = threadIdx.x;
    float a = 0.0f, b = 0.0f;
    for (int i = blockIdx.x * blockDim.x + tid; i < E; i += gridDim.x * blockDim.x) {
        float v = ew[i]; a += v; b += v * v;
    }
    s1[tid] = a; s2[tid] = b; __syncthreads();
    for (int ofs = 128; ofs > 0; ofs >>= 1) {
        if (tid < ofs) { s1[tid] += s1[tid + ofs]; s2[tid] += s2[tid + ofs]; }
        __syncthreads();
    }
    if (tid == 0) { atomicAdd(&stats[0], s1[0]); atomicAdd(&stats[1], s2[0]); }
}

// normalize ew to mean-0 var-1e-4 +1 (unbiased var) and accumulate degree
__global__ void ew_norm_deg_kernel(float* __restrict__ ew,
                                   const float* __restrict__ stats,
                                   const int* __restrict__ row,
                                   float* __restrict__ deg, int E) {
    int e = blockIdx.x * blockDim.x + threadIdx.x;
    if (e >= E) return;
    float sum = stats[0], sumsq = stats[1];
    float mean = sum / (float)E;
    float var  = (sumsq - sum * sum / (float)E) / (float)(E - 1);
    float sc   = sqrtf(1e-4f / var);
    float v = (ew[e] - mean) * sc + 1.0f;
    ew[e] = v;
    atomicAdd(&deg[row[e]], v);                    // deg pre-filled with 1 (self loop)
}

__global__ void node_norm_kernel(const float* __restrict__ deg,
                                 float* __restrict__ dinv,
                                 float* __restrict__ wl,
                                 float* __restrict__ degM, int n) {
    int i = blockIdx.x * blockDim.x + threadIdx.x;
    if (i >= n) return;
    float d  = deg[i];
    float di = d > 0.0f ? rsqrtf(fmaxf(d, 1e-12f)) : 0.0f;
    dinv[i] = di;
    float w = di * di;                             // self-loop weight after norm
    wl[i]   = w;
    degM[i] = w;                                   // init degM with self loop
}

__global__ void edge_weight_kernel(const float* __restrict__ ewn,
                                   const float* __restrict__ dinv,
                                   const int* __restrict__ row,
                                   const int* __restrict__ col,
                                   float* __restrict__ wE,
                                   float* __restrict__ degM, int E) {
    int e = blockIdx.x * blockDim.x + threadIdx.x;
    if (e >= E) return;
    int r = row[e], c = col[e];
    float w = dinv[r] * ewn[e] * dinv[c];
    wE[e] = w;
    atomicAdd(&degM[r], w);
}

__global__ void alpha_beta_kernel(const float* __restrict__ degM,
                                  float* __restrict__ alpha,
                                  float* __restrict__ beta, int n) {
    int i = blockIdx.x * blockDim.x + threadIdx.x;
    if (i >= n) return;
    float a = 1.0f / (degM[i] + MU_C);
    alpha[i] = a;
    beta[i]  = MU_C * a;
}

// --------------------------------------------------------- sparse aggregate
// agg[i,:] = wl[i] * src[i,:]   (self-loop term, also zero-initializes agg)
__global__ void agg_init_kernel(const float* __restrict__ src,
                                const float* __restrict__ wl,
                                float* __restrict__ agg, long n, int dshift) {
    long idx = (long)blockIdx.x * blockDim.x + threadIdx.x;
    if (idx >= n) return;
    agg[idx] = wl[idx >> dshift] * src[idx];
}

// agg[row[e],:] += wE[e] * src[col[e],:]  — 4 floats per thread, atomics
__global__ void spmm_edge_kernel(const float* __restrict__ src,
                                 const float* __restrict__ wE,
                                 const int* __restrict__ row,
                                 const int* __restrict__ col,
                                 float* __restrict__ agg,
                                 long total, int cshift, int D) {
    long idx = (long)blockIdx.x * blockDim.x + threadIdx.x;
    if (idx >= total) return;
    int e  = (int)(idx >> cshift);
    int d4 = (int)(idx & ((1 << cshift) - 1)) << 2;
    float w = wE[e];
    const float* s = src + (long)col[e] * D + d4;
    float* a = agg + (long)row[e] * D + d4;
    float4 v = *(const float4*)s;
    atomicAdd(a + 0, w * v.x);
    atomicAdd(a + 1, w * v.y);
    atomicAdd(a + 2, w * v.z);
    atomicAdd(a + 3, w * v.w);
}

// out = alpha*agg + beta*xlin; optional relu->f16 copy (feeds next GEMM)
__global__ void combine_kernel(const float* __restrict__ agg,
                               const float* __restrict__ xlin,
                               const float* __restrict__ alpha,
                               const float* __restrict__ beta,
                               float* __restrict__ outF,
                               _Float16* __restrict__ outH,
                               long n, int dshift) {
    long idx = (long)blockIdx.x * blockDim.x + threadIdx.x;
    if (idx >= n) return;
    long i = idx >> dshift;
    float v = alpha[i] * agg[idx] + beta[i] * xlin[idx];
    outF[idx] = v;
    if (outH) outH[idx] = (_Float16)fmaxf(v, 0.0f);
}

// ---------------------------------------------------------------------------
extern "C" void kernel_launch(void* const* d_in, const int* in_sizes, int n_in,
                              void* d_out, int out_size, void* d_ws, size_t ws_size,
                              hipStream_t stream) {
    const float* x        = (const float*)d_in[0];
    const int*   eidx     = (const int*)  d_in[1];
    const float* lin1_W   = (const float*)d_in[2];
    const float* lin1_b   = (const float*)d_in[3];
    const float* mlp_W1   = (const float*)d_in[4];
    const float* mlp_b1   = (const float*)d_in[5];
    const float* mlp_W2   = (const float*)d_in[6];
    const float* mlp_b2   = (const float*)d_in[7];
    const float* mlp_W3   = (const float*)d_in[8];
    const float* mlp_b3   = (const float*)d_in[9];
    const float* parsing0 = (const float*)d_in[10];
    const float* conv0_W  = (const float*)d_in[11];
    const float* conv0_b  = (const float*)d_in[12];
    const float* conv1_W  = (const float*)d_in[13];
    const float* conv1_b  = (const float*)d_in[14];
    const int* row = eidx;
    const int* col = eidx + NUM_EDGES;
    float* out_final = (float*)d_out;

    const int  N = NUM_NODES, E = NUM_EDGES;
    const long Nf512 = (long)N * 512, Nf128 = (long)N * 128;

    // ----- workspace carving (256B aligned) -----
    char* ws = (char*)d_ws; size_t off = 0;
    auto alloc = [&](size_t bytes) -> void* {
        void* p = ws + off; off += (bytes + 255) & ~(size_t)255; return p;
    };
    _Float16* xh      = (_Float16*)alloc(Nf512 * 2);       // x in f16
    _Float16* h1      = (_Float16*)alloc(Nf512 * 2);       // mlp hidden1 (reused as zh)
    _Float16* h2      = (_Float16*)alloc((long)N * 64 * 2);
    float*    logits  = (float*)   alloc(Nf128 * 4);
    _Float16* logitsh = (_Float16*)alloc(Nf128 * 2);
    float*    tP      = (float*)   alloc(Nf128 * 4);
    float*    ew      = (float*)   alloc((long)E * 4);
    float*    stats   = (float*)   alloc(256);
    float*    deg     = (float*)   alloc((long)N * 4);
    float*    dinv    = (float*)   alloc((long)N * 4);
    float*    wl      = (float*)   alloc((long)N * 4);
    float*    degM    = (float*)   alloc((long)N * 4);
    float*    alpha   = (float*)   alloc((long)N * 4);
    float*    beta    = (float*)   alloc((long)N * 4);
    float*    wE      = (float*)   alloc((long)E * 4);
    _Float16* W1t     = (_Float16*)alloc(512 * 512 * 2);
    _Float16* W2t     = (_Float16*)alloc(512 * 64 * 2);
    _Float16* W3t     = (_Float16*)alloc(64 * 128 * 2);
    _Float16* Pt      = (_Float16*)alloc(128 * 128 * 2);
    _Float16* lin1t   = (_Float16*)alloc(512 * 512 * 2);
    _Float16* conv0t  = (_Float16*)alloc(512 * 512 * 2);
    _Float16* conv1t  = (_Float16*)alloc(512 * 128 * 2);
    float*    xlin0   = (float*)   alloc(Nf512 * 4);
    float*    out0    = (float*)   alloc(Nf512 * 4);
    float*    agg0    = (float*)   alloc(Nf512 * 4);       // layer1 bufs alias into here
    // safe aliases (dead-after-use regions)
    _Float16* zh    = h1;                                  // h1 dead after mlp2 GEMM
    _Float16* outrh = xh;                                  // xh dead after lin1/mlp1 GEMMs
    float* xlin1 = agg0;                                   // agg0 region: 40.96MB >= 3*10.24MB
    float* out1  = (float*)((char*)agg0 + ((Nf128 * 4 + 255) & ~(size_t)255));
    float* agg1  = (float*)((char*)out1 + ((Nf128 * 4 + 255) & ~(size_t)255));

    auto cdiv = [](long a, long b) { return (int)((a + b - 1) / b); };
    auto gemm = [&](const _Float16* A, const _Float16* Bt, const float* bias,
                    float* Cf, _Float16* Ch, int M, int K, int Nc, int relu) {
        int waves = ((M >> 4) / MB) * ((Nc >> 4) / NB);
        gemm_f16_wmma_kernel<<<cdiv(waves, 8), 256, 0, stream>>>(
            A, Bt, bias, Cf, Ch, M, K, Nc, relu);
    };

    // ----- stage 0: conversions / weight transposes -----
    f32_to_f16_kernel<<<cdiv(Nf512, 256), 256, 0, stream>>>(x, xh, Nf512);
    transpose_w_kernel<<<cdiv(512*512,256),256,0,stream>>>(mlp_W1, W1t, 512, 512, 1.f, 0);
    transpose_w_kernel<<<cdiv(512*64 ,256),256,0,stream>>>(mlp_W2, W2t, 512, 64,  1.f, 0);
    transpose_w_kernel<<<cdiv(64*128 ,256),256,0,stream>>>(mlp_W3, W3t, 64,  128, 1.f, 0);
    transpose_w_kernel<<<cdiv(128*128,256),256,0,stream>>>(parsing0, Pt, 128, 128, 2.f, 1); // relu(SCALE*P0)
    transpose_w_kernel<<<cdiv(512*512,256),256,0,stream>>>(lin1_W, lin1t, 512, 512, 1.f, 0);
    transpose_w_kernel<<<cdiv(512*512,256),256,0,stream>>>(conv0_W, conv0t, 512, 512, 1.f, 0);
    transpose_w_kernel<<<cdiv(512*128,256),256,0,stream>>>(conv1_W, conv1t, 512, 128, 1.f, 0);

    // ----- stage 1: MLP for soft edge logits -----
    gemm(xh, W1t, mlp_b1, nullptr, h1, N, 512, 512, 1);        // h1 = relu(x@W1+b1)
    gemm(h1, W2t, mlp_b2, nullptr, h2, N, 512, 64, 1);         // h2 = relu(h1@W2+b2)
    gemm(xh, lin1t, lin1_b, nullptr, zh, N, 512, 512, 0);      // zh = x@lin1+b (aliases h1, now dead)
    gemm(h2, W3t, mlp_b3, logits, logitsh, N, 64, 128, 0);     // logits
    gemm(logitsh, Pt, nullptr, tP, nullptr, N, 128, 128, 0);   // tP = logits@parsing

    // ----- stage 2: edge weights + normalization -----
    edge_dot_kernel<<<cdiv(E, 256), 256, 0, stream>>>(logits, tP, row, col, ew, E);
    fill_f32_kernel<<<1, 64, 0, stream>>>(stats, 0.0f, 2);
    reduce_stats_kernel<<<240, 256, 0, stream>>>(ew, stats, E);
    fill_f32_kernel<<<cdiv(N, 256), 256, 0, stream>>>(deg, 1.0f, N);   // self-loop weight 1
    ew_norm_deg_kernel<<<cdiv(E, 256), 256, 0, stream>>>(ew, stats, row, deg, E);
    node_norm_kernel<<<cdiv(N, 256), 256, 0, stream>>>(deg, dinv, wl, degM, N);
    edge_weight_kernel<<<cdiv(E, 256), 256, 0, stream>>>(ew, dinv, row, col, wE, degM, E);
    alpha_beta_kernel<<<cdiv(N, 256), 256, 0, stream>>>(degM, alpha, beta, N);

    // ----- stage 3: pGNN layer 0 (hidden->hidden, D=512, K=2 iters) -----
    gemm(zh, conv0t, conv0_b, xlin0, nullptr, N, 512, 512, 0); // xlin0 = z@conv0+b
    {
        const long tot = (long)E * (512 / 4);
        for (int it = 0; it < 2; ++it) {
            const float* src = (it == 0) ? xlin0 : out0;
            agg_init_kernel<<<cdiv(Nf512, 256), 256, 0, stream>>>(src, wl, agg0, Nf512, 9);
            spmm_edge_kernel<<<cdiv(tot, 256), 256, 0, stream>>>(src, wE, row, col, agg0, tot, 7, 512);
            combine_kernel<<<cdiv(Nf512, 256), 256, 0, stream>>>(
                agg0, xlin0, alpha, beta, out0,
                (it == 1) ? outrh : (_Float16*)nullptr, Nf512, 9); // final: relu->f16
        }
    }

    // ----- stage 4: pGNN layer 1 (hidden->out, D=128, K=2 iters) -----
    gemm(outrh, conv1t, conv1_b, xlin1, nullptr, N, 512, 128, 0);
    {
        const long tot = (long)E * (128 / 4);
        for (int it = 0; it < 2; ++it) {
            const float* src = (it == 0) ? xlin1 : out1;
            agg_init_kernel<<<cdiv(Nf128, 256), 256, 0, stream>>>(src, wl, agg1, Nf128, 7);
            spmm_edge_kernel<<<cdiv(tot, 256), 256, 0, stream>>>(src, wE, row, col, agg1, tot, 5, 128);
            combine_kernel<<<cdiv(Nf128, 256), 256, 0, stream>>>(
                agg1, xlin1, alpha, beta,
                (it == 1) ? out_final : out1, (_Float16*)nullptr, Nf128, 7);
        }
    }
}
